// SingleHeadAttention_89472758710814
// MI455X (gfx1250) — matrix operand order repaired and checked by reading, not compiled, not run on hardware
//
#include <hip/hip_runtime.h>

typedef __attribute__((ext_vector_type(16))) __bf16 v16bf;
typedef __attribute__((ext_vector_type(2)))  __bf16 v2bf;
typedef __attribute__((ext_vector_type(8)))  float  v8f;
typedef __attribute__((ext_vector_type(2)))  float  v2f;

#define T_SEQ   2048
#define HDIM    128
#define BQ      128      // query rows per workgroup
#define BK      64       // keys per iteration
#define NWAVES  8

// LDS strides in bf16 elements (padded to dodge bank conflicts)
#define K_STR   136      // K tile rows: 64 keys x 128 dims
#define VT_STR  72       // V^T rows: 128 dims x 64 keys
#define P_STR   72       // P scratch rows: 16 x 64 per wave

// log2(e) / sqrt(128): folds softmax scale AND base-2 exp into Q.
// softmax_e(qk/sqrt(H)) == exp2-softmax of (q*QSCALE)k, exactly.
#define QSCALE  0.1275188297384295f

__device__ __forceinline__ unsigned pk2bf(float a, float b) {
    v2f  f = {a, b};
    v2bf h = __builtin_convertvector(f, v2bf);   // packed f32->bf16 convert
    return __builtin_bit_cast(unsigned, h);
}
__device__ __forceinline__ __bf16 f2bf(float x) {
    return (__bf16)x;
}

__global__ __launch_bounds__(256) void
fa_fwd_causal(const float* __restrict__ Q, const float* __restrict__ K,
              const float* __restrict__ V, float* __restrict__ O)
{
    __shared__ __align__(16) __bf16 sK [2][BK * K_STR];          // 2 x 17408 B
    __shared__ __align__(16) __bf16 sVt[2][HDIM * VT_STR];       // 2 x 18432 B
    __shared__ __align__(16) __bf16 sP [NWAVES * 16 * P_STR];    // 18432 B

    const int qblk = blockIdx.x;
    const int b    = blockIdx.y;
    const int tid  = threadIdx.x;
    const int wave = tid >> 5;
    const int lane = tid & 31;
    const int half = lane >> 4;
    const int l16  = lane & 15;

    const float* Qb = Q + (size_t)b * T_SEQ * HDIM;
    const float* Kb = K + (size_t)b * T_SEQ * HDIM;
    const float* Vb = V + (size_t)b * T_SEQ * HDIM;
    float*       Ob = O + (size_t)b * T_SEQ * HDIM;

    const int qr0 = qblk * BQ + wave * 16;        // first query row of this wave

    // ---- staging work distribution (fixed per thread) ----
    const int st_kr  = tid & 63;          // K: key row, 32 contiguous dims
    const int st_kc0 = (tid >> 6) * 32;
    const int st_vk0 = (tid & 31) * 2;    // V: key pair, 16 contiguous dims
    const int st_vd0 = (tid >> 5) * 16;

    float kst[32], vst0[16], vst1[16];    // register prefetch buffers

    auto load_regs = [&](int kv0) {       // issue global loads (latency overlapped)
        const float4* kr4 = reinterpret_cast<const float4*>(
                                Kb + (size_t)(kv0 + st_kr) * HDIM + st_kc0);
        #pragma unroll
        for (int i = 0; i < 8; ++i) {
            const float4 t = kr4[i];
            kst[4*i+0] = t.x; kst[4*i+1] = t.y; kst[4*i+2] = t.z; kst[4*i+3] = t.w;
        }
        const float4* v0 = reinterpret_cast<const float4*>(
                                Vb + (size_t)(kv0 + st_vk0) * HDIM + st_vd0);
        const float4* v1 = reinterpret_cast<const float4*>(
                                Vb + (size_t)(kv0 + st_vk0 + 1) * HDIM + st_vd0);
        #pragma unroll
        for (int i = 0; i < 4; ++i) {
            const float4 a = v0[i];
            vst0[4*i+0] = a.x; vst0[4*i+1] = a.y; vst0[4*i+2] = a.z; vst0[4*i+3] = a.w;
            const float4 c = v1[i];
            vst1[4*i+0] = c.x; vst1[4*i+1] = c.y; vst1[4*i+2] = c.z; vst1[4*i+3] = c.w;
        }
    };
    auto store_lds = [&](int buf) {       // convert fp32 -> packed bf16 -> LDS
        unsigned* dk = reinterpret_cast<unsigned*>(
                           &sK[buf][(size_t)st_kr * K_STR + st_kc0]);
        #pragma unroll
        for (int j = 0; j < 16; ++j) dk[j] = pk2bf(kst[2*j], kst[2*j+1]);
        unsigned* dv = reinterpret_cast<unsigned*>(&sVt[buf][0]);
        #pragma unroll
        for (int i = 0; i < 16; ++i)
            dv[(size_t)(st_vd0 + i) * (VT_STR / 2) + (st_vk0 >> 1)]
                = pk2bf(vst0[i], vst1[i]);
    };

    // ---- load Q (pre-scaled) as 4 bf16 A-fragments (16x32), covering H=128 ----
    // A layout: M = lane%16; elem e: K = 16*half + e (contiguous)
    v16bf aq[4];
    {
        const float* qrow = Qb + (size_t)(qr0 + l16) * HDIM;
        #pragma unroll
        for (int c = 0; c < 4; ++c) {
            #pragma unroll
            for (int e = 0; e < 16; ++e) {
                const int v = e >> 1, p = e & 1;
                const int kk = half * 16 + ((v << 1) + p);
                aq[c][e] = f2bf(qrow[c * 32 + kk] * QSCALE);
            }
        }
    }

    // all-ones B fragment: row-sum of P via WMMA (P * ones -> l-layout row sums)
    v16bf bones;
    #pragma unroll
    for (int e = 0; e < 16; ++e) bones[e] = f2bf(1.0f);

    // ---- online-softmax state: rows M = r + 8*half, N = l16 (C-layout) ----
    float m_i[8], l_i[8];
    v8f   o_acc[8];
    #pragma unroll
    for (int r = 0; r < 8; ++r) { m_i[r] = -1.0e30f; l_i[r] = 0.0f; }
    #pragma unroll
    for (int n = 0; n < 8; ++n) o_acc[n] = (v8f){};

    const int kv_end = qblk * BQ + BQ;

    // ---- software pipeline prologue: block 0 into buffer 0 ----
    load_regs(0);
    store_lds(0);
    __syncthreads();

    int buf = 0;
    for (int kv0 = 0; kv0 < kv_end; kv0 += BK) {
        const bool has_next = (kv0 + BK) < kv_end;   // workgroup-uniform
        if (has_next) load_regs(kv0 + BK);           // prefetch next block

        if (kv0 <= qr0 + 15) {                       // wave has >=1 unmasked element
            const __bf16* sKb  = &sK [buf][0];
            const __bf16* sVtb = &sVt[buf][0];

            // ---- S = Q * K^T : ch outer / t inner -> 4 independent WMMA chains ----
            v8f s[4];
            #pragma unroll
            for (int t = 0; t < 4; ++t) s[t] = (v8f){};
            #pragma unroll
            for (int ch = 0; ch < 4; ++ch) {
                v16bf bk[4];
                #pragma unroll
                for (int t = 0; t < 4; ++t) {
                    // B layout: N = l16 (key in tile), K-dim = 16*half + e -> contiguous
                    const __bf16* krow = sKb + (t * 16 + l16) * K_STR + ch * 32 + half * 16;
                    #pragma unroll
                    for (int e = 0; e < 16; ++e) bk[t][e] = krow[e];
                }
                #pragma unroll
                for (int t = 0; t < 4; ++t)
                    s[t] = __builtin_amdgcn_wmma_f32_16x16x32_bf16(
                               false, aq[ch], false, bk[t], (short)0, s[t], false, false);
            }

            // ---- causal mask only on diagonal blocks (wave-uniform branch) ----
            if (kv0 + BK - 1 > qr0) {
                #pragma unroll
                for (int t = 0; t < 4; ++t)
                    #pragma unroll
                    for (int r = 0; r < 8; ++r)
                        if (kv0 + t * 16 + l16 > qr0 + r + 8 * half)
                            s[t][r] = -1.0e30f;
            }

            // ---- online softmax (base-2 domain); p=exp2 fused with P->LDS store ----
            __bf16* pw = sP + wave * 16 * P_STR;
            float alpha[8];
            #pragma unroll
            for (int r = 0; r < 8; ++r) {
                float v = fmaxf(fmaxf(s[0][r], s[1][r]), fmaxf(s[2][r], s[3][r]));
                v = fmaxf(v, __shfl_xor(v, 1, 32));
                v = fmaxf(v, __shfl_xor(v, 2, 32));
                v = fmaxf(v, __shfl_xor(v, 4, 32));
                v = fmaxf(v, __shfl_xor(v, 8, 32));
                const float mnew = fmaxf(m_i[r], v);
                alpha[r] = __builtin_amdgcn_exp2f(m_i[r] - mnew);
                m_i[r]   = mnew;
                #pragma unroll
                for (int t = 0; t < 4; ++t) {
                    const float p = __builtin_amdgcn_exp2f(s[t][r] - mnew);
                    pw[(r + 8 * half) * P_STR + t * 16 + l16] = f2bf(p);
                }
            }
            #pragma unroll
            for (int n = 0; n < 8; ++n)
                #pragma unroll
                for (int r = 0; r < 8; ++r)
                    o_acc[n][r] *= alpha[r];

            // ---- read P back as two A-layout fragments ----
            v16bf ap[2];
            {
                const __bf16* prow = pw + l16 * P_STR;
                #pragma unroll
                for (int kc = 0; kc < 2; ++kc) {
                    #pragma unroll
                    for (int e = 0; e < 8; ++e) ap[kc][e]     = prow[kc * 32 + half * 16 + e];
                    #pragma unroll
                    for (int e = 0; e < 8; ++e) ap[kc][8 + e] = prow[kc * 32 + half * 16 + 8 + e];
                }
            }

            // ---- row sums of P via WMMA with ones-B ----
            v8f csum = (v8f){};
            csum = __builtin_amdgcn_wmma_f32_16x16x32_bf16(
                       false, ap[0], false, bones, (short)0, csum, false, false);
            csum = __builtin_amdgcn_wmma_f32_16x16x32_bf16(
                       false, ap[1], false, bones, (short)0, csum, false, false);
            #pragma unroll
            for (int r = 0; r < 8; ++r)
                l_i[r] = fmaf(l_i[r], alpha[r], csum[r]);

            // ---- O += P * V : kc outer / n inner -> 8 independent WMMA chains ----
            #pragma unroll
            for (int kc = 0; kc < 2; ++kc) {
                #pragma unroll
                for (int n = 0; n < 8; ++n) {
                    // B layout: N(dim) = n*16+l16, K(key) = kc*32 + 16*half + e -> contiguous
                    v16bf bv;
                    const __bf16* vcol = sVtb + (size_t)(n * 16 + l16) * VT_STR
                                              + kc * 32 + half * 16;
                    #pragma unroll
                    for (int e = 0; e < 16; ++e) bv[e] = vcol[e];
                    o_acc[n] = __builtin_amdgcn_wmma_f32_16x16x32_bf16(
                                   false, ap[kc], false, bv, (short)0, o_acc[n], false, false);
                }
            }
        }

        if (has_next) store_lds(buf ^ 1);    // fill other buffer (waits loads here)
        __syncthreads();                     // single barrier per iteration
        buf ^= 1;
    }

    // ---- epilogue: normalize and store fp32 ----
    float inv_l[8];
    #pragma unroll
    for (int r = 0; r < 8; ++r) inv_l[r] = 1.0f / l_i[r];

    #pragma unroll
    for (int n = 0; n < 8; ++n)
        #pragma unroll
        for (int r = 0; r < 8; ++r)
            Ob[(size_t)(qr0 + r + 8 * half) * HDIM + n * 16 + l16] = o_acc[n][r] * inv_l[r];
}

extern "C" void kernel_launch(void* const* d_in, const int* in_sizes, int n_in,
                              void* d_out, int out_size, void* d_ws, size_t ws_size,
                              hipStream_t stream) {
    const float* q = (const float*)d_in[0];
    const float* k = (const float*)d_in[1];
    const float* v = (const float*)d_in[2];
    // d_in[3] is the causal mask; causality is applied analytically in-kernel.
    float* out = (float*)d_out;

    const int B = in_sizes[0] / (T_SEQ * HDIM);   // 16
    dim3 grid(T_SEQ / BQ, B);
    dim3 block(256);
    fa_fwd_causal<<<grid, block, 0, stream>>>(q, k, v, out);
}
